// kWayLayer_74912819577345
// MI455X (gfx1250) — matrix operand order repaired and checked by reading, not compiled, not run on hardware
//
#include <hip/hip_runtime.h>
#include <stdint.h>

#define N_PAIRS   50000
#define NUM_E     38
#define F         1140
#define FPAD      1152            // padded K/J, multiple of 32
#define M_BLK     128             // rows (pairs) per block
#define A_STRIDE  1160            // FPAD + 8 elements: 4-bank rotation per row, conflict-free b128 reads
#define W_STRIDE  136             // 128 + 8 elements: same trick for transposed W chunk
#define JT        64              // j columns per outer iteration (2 j-wave groups x 2 subtiles)
#define KCHUNK    128             // K elements staged per barrier round
#define NCHUNK    (FPAD / KCHUNK) // 9

typedef __attribute__((ext_vector_type(16))) __bf16 bf16x16;
typedef __attribute__((ext_vector_type(8)))  __bf16 bf16x8;
typedef __attribute__((ext_vector_type(4)))  __bf16 bf16x4;
typedef __attribute__((ext_vector_type(2)))  __bf16 bf16x2;
typedef __attribute__((ext_vector_type(8)))  float  f32x8;

// Load one 128(i) x 64(j) chunk of W_e into registers as (i, i+1) pairs at fixed j.
// Each wave's lanes sweep 32 consecutive j -> fully coalesced 128B b32 load groups.
__device__ __forceinline__ void wload(float* wr, const float* __restrict__ wbase,
                                      int k0, int jg, int ilb, bool full) {
    if (full) {
#pragma unroll
        for (int s = 0; s < 16; ++s) {
            const int ig = k0 + ilb + s * 8;
            const float* p = wbase + (size_t)ig * F + jg;
            wr[2 * s]     = p[0];
            wr[2 * s + 1] = p[F];
        }
    } else {
        const bool jok = (jg < F);
#pragma unroll
        for (int s = 0; s < 16; ++s) {
            const int ig = k0 + ilb + s * 8;
            wr[2 * s]     = (jok && ig < F)       ? wbase[(size_t)ig * F + jg]       : 0.0f;
            wr[2 * s + 1] = (jok && (ig + 1) < F) ? wbase[(size_t)(ig + 1) * F + jg] : 0.0f;
        }
    }
}

// Pack register pairs to bf16, store transposed chunk: sW[jj][i], one b32 per pair.
__device__ __forceinline__ void wstore(const float* wr, __bf16* sW, int jj, int ilb) {
#pragma unroll
    for (int s = 0; s < 16; ++s) {
        bf16x2 p;
        p.x = (__bf16)wr[2 * s];
        p.y = (__bf16)wr[2 * s + 1];
        *(bf16x2*)((void*)&sW[jj * W_STRIDE + ilb + s * 8]) = p;
    }
}

__device__ __forceinline__ bf16x16 ldsA(const __bf16* p) {
    bf16x16 a;
    ((bf16x8*)&a)[0] = *(const bf16x8*)((const void*)p);
    ((bf16x8*)&a)[1] = *(const bf16x8*)((const void*)(p + 16));
    return a;
}
__device__ __forceinline__ bf16x16 ldsB(const __bf16* p) {
    bf16x16 b;
    ((bf16x8*)&b)[0] = *(const bf16x8*)((const void*)p);
    ((bf16x8*)&b)[1] = *(const bf16x8*)((const void*)(p + 16));
    return b;
}

__global__ __launch_bounds__(256)
void kway_bilinear_wmma(const float* __restrict__ start_emb,
                        const float* __restrict__ end_emb,
                        const float* __restrict__ weights,
                        float* __restrict__ out)
{
    // 296,960 B A-tile + 17,408 B W-chunk = 314,368 B < 320 KB/WGP
    __shared__ __bf16 sA[M_BLK * A_STRIDE];
    __shared__ __bf16 sW[JT * W_STRIDE];

    const int tid  = threadIdx.x;
    const int e    = blockIdx.y;
    const int row0 = blockIdx.x * M_BLK;
    const float* wbase = weights + (size_t)e * F * F;

    // ---- Stage A (start rows) once: fp32 -> bf16, zero-padded rows/cols ----
    {
        const int ROWV4 = F / 4;                       // 285 float4 per row
        for (int idx = tid; idx < M_BLK * ROWV4; idx += 256) {
            int rl  = idx / ROWV4;
            int c4  = idx - rl * ROWV4;
            int row = row0 + rl;
            float4 v = make_float4(0.f, 0.f, 0.f, 0.f);
            if (row < N_PAIRS)
                v = *(const float4*)(start_emb + (size_t)row * F + (size_t)c4 * 4);
            bf16x4 p;
            p.x = (__bf16)v.x; p.y = (__bf16)v.y; p.z = (__bf16)v.z; p.w = (__bf16)v.w;
            *(bf16x4*)((void*)&sA[rl * A_STRIDE + c4 * 4]) = p;   // 8B aligned
        }
        for (int idx = tid; idx < M_BLK * (FPAD - F); idx += 256) {
            int rl = idx / (FPAD - F);
            int c  = F + (idx - rl * (FPAD - F));
            sA[rl * A_STRIDE + c] = (__bf16)0.0f;
        }
    }

    const int lane = tid & 31;
    const int n16  = lane & 15;        // WMMA tile column / A row-in-strip
    const int half = lane >> 4;        // lane half: K sub-range select / C row half
    const int wave = tid >> 5;
    const int rw   = wave & 3;         // row group: rows rw*32 .. rw*32+31 (two 16-row tiles)
    const int jw   = wave >> 2;        // j group: columns jw*32 .. jw*32+31 (two 16-col subtiles)

    // staging decomposition for W chunks (64 j columns x 128 i rows)
    const int jj  = tid & 63;
    const int ilb = (tid >> 6) * 2;

    float acc0[8], acc1[8];            // per-row partial dots, tiles 0 (rows+0..15) / 1 (rows+16..31)
#pragma unroll
    for (int r = 0; r < 8; ++r) { acc0[r] = 0.0f; acc1[r] = 0.0f; }

    for (int j0 = 0; j0 < FPAD; j0 += JT) {
        const bool fullj = (j0 + JT <= F);
        const int  jg    = j0 + jj;
        f32x8 c00 = {}, c01 = {}, c10 = {}, c11 = {};

        float wr[32];
        wload(wr, wbase, 0, jg, ilb, fullj);     // chunk 0 (k0=0 always K-full)

        for (int c = 0; c < NCHUNK; ++c) {
            const int k0 = c * KCHUNK;
            __syncthreads();                     // previous chunk's LDS readers done
            wstore(wr, sW, jj, ilb);
            if (c + 1 < NCHUNK)                  // pipeline next chunk's global loads under compute
                wload(wr, wbase, k0 + KCHUNK, jg, ilb,
                      fullj && (k0 + 2 * KCHUNK <= F));
            __syncthreads();                     // sW ready

#pragma unroll
            for (int kk = 0; kk < KCHUNK; kk += 32) {
                const int ko = k0 + kk;
                // Two A row-tiles (registers reused across both B subtiles)
                bf16x16 a0 = ldsA(&sA[(rw * 32 + n16)      * A_STRIDE + ko + half * 8]);
                bf16x16 a1 = ldsA(&sA[(rw * 32 + 16 + n16) * A_STRIDE + ko + half * 8]);
                // Two B column-subtiles (reused across both A tiles)
                bf16x16 b0 = ldsB(&sW[(jw * 32 + n16)      * W_STRIDE + kk + half * 8]);
                bf16x16 b1 = ldsB(&sW[(jw * 32 + 16 + n16) * W_STRIDE + kk + half * 8]);
                c00 = __builtin_amdgcn_wmma_f32_16x16x32_bf16(false, a0, false, b0, (short)0, c00, false, false);
                c01 = __builtin_amdgcn_wmma_f32_16x16x32_bf16(false, a0, false, b1, (short)0, c01, false, false);
                c10 = __builtin_amdgcn_wmma_f32_16x16x32_bf16(false, a1, false, b0, (short)0, c10, false, false);
                c11 = __builtin_amdgcn_wmma_f32_16x16x32_bf16(false, a1, false, b1, (short)0, c11, false, false);
            }
        }

        // Fold proj tiles into per-row dot with end[row, j].
        // C layout: VGPR r -> tile row half*8+r, lane -> column n16.
        const int ja = j0 + jw * 32 + n16;
        const int jb = ja + 16;
#pragma unroll
        for (int r = 0; r < 8; ++r) {
            const int r0 = row0 + rw * 32 + half * 8 + r;
            if (r0 < N_PAIRS) {
                const float* ep = end_emb + (size_t)r0 * F;
                if (ja < F) acc0[r] += c00[r] * ep[ja];
                if (jb < F) acc0[r] += c01[r] * ep[jb];
            }
            const int r1 = r0 + 16;
            if (r1 < N_PAIRS) {
                const float* ep = end_emb + (size_t)r1 * F;
                if (ja < F) acc1[r] += c10[r] * ep[ja];
                if (jb < F) acc1[r] += c11[r] * ep[jb];
            }
        }
    }

    // Reduce across the 16 column-lanes (masks < 16 stay within each half)
#pragma unroll
    for (int r = 0; r < 8; ++r) {
        float v0 = acc0[r], v1 = acc1[r];
        v0 += __shfl_xor(v0, 8); v1 += __shfl_xor(v1, 8);
        v0 += __shfl_xor(v0, 4); v1 += __shfl_xor(v1, 4);
        v0 += __shfl_xor(v0, 2); v1 += __shfl_xor(v1, 2);
        v0 += __shfl_xor(v0, 1); v1 += __shfl_xor(v1, 1);
        acc0[r] = v0; acc1[r] = v1;
    }

    // Cross-j-wave combine via LDS scratch (alias sW after a barrier), then single store per row.
    float* sRed = (float*)sW;
    __syncthreads();
    if (jw == 0 && n16 == 0) {
#pragma unroll
        for (int r = 0; r < 8; ++r) {
            sRed[rw * 32 + half * 8 + r]      = acc0[r];
            sRed[rw * 32 + 16 + half * 8 + r] = acc1[r];
        }
    }
    __syncthreads();
    if (jw == 1 && n16 == 0) {
#pragma unroll
        for (int r = 0; r < 8; ++r) {
            const int lr0 = rw * 32 + half * 8 + r;
            const int lr1 = lr0 + 16;
            const int r0  = row0 + lr0;
            const int r1  = row0 + lr1;
            if (r0 < N_PAIRS) out[(size_t)e * N_PAIRS + r0] = sRed[lr0] + acc0[r];
            if (r1 < N_PAIRS) out[(size_t)e * N_PAIRS + r1] = sRed[lr1] + acc1[r];
        }
    }
}

extern "C" void kernel_launch(void* const* d_in, const int* in_sizes, int n_in,
                              void* d_out, int out_size, void* d_ws, size_t ws_size,
                              hipStream_t stream) {
    (void)in_sizes; (void)n_in; (void)d_ws; (void)ws_size; (void)out_size;
    const float* start_emb = (const float*)d_in[0];
    const float* end_emb   = (const float*)d_in[1];
    const float* weights   = (const float*)d_in[2];
    float* out             = (float*)d_out;

    dim3 grid((N_PAIRS + M_BLK - 1) / M_BLK, NUM_E);   // x-fastest: all blocks of one e together -> W_e stays in L2
    dim3 block(256);
    kway_bilinear_wmma<<<grid, block, 0, stream>>>(start_emb, end_emb, weights, out);
}
